// EnhancedCrossAttention_71339406786729
// MI455X (gfx1250) — matrix operand order repaired and checked by reading, not compiled, not run on hardware
//
#include <hip/hip_runtime.h>

// ---------------------------------------------------------------------------
// EnhancedCrossAttention for MI455X (gfx1250, wave32, WMMA + TDM/async-LDS)
// B=4, C=256, H=W=28 -> S = 2*784 = 1568, heads=8, dh=32
// Matmuls in f16 via v_wmma_f32_16x16x32_f16, f32 accumulate.
// Attention K tiles staged with tensor_load_to_lds (TDM), V tiles with
// global_load_async_to_lds_b128, double-buffered in LDS.
// ---------------------------------------------------------------------------

typedef __attribute__((ext_vector_type(16))) _Float16 v16h;
typedef __attribute__((ext_vector_type(8)))  _Float16 v8h;
typedef __attribute__((ext_vector_type(8)))  float    v8f;
typedef unsigned int u32x4 __attribute__((ext_vector_type(4)));
typedef int          i32x4 __attribute__((ext_vector_type(4)));
typedef int          i32x8 __attribute__((ext_vector_type(8)));

#define DEV __device__ __forceinline__

constexpr int Bc = 4;
constexpr int Cc = 256;
constexpr int Hc = 28;
constexpr int Wc = 28;
constexpr int Np = Hc * Wc;        // 784 pixels
constexpr int Sc = 2 * Np;         // 1568 sequence length
constexpr int NH = 8;
constexpr int DH = 32;
constexpr int C3 = 3 * Cc;         // 768
constexpr float ATTN_SCALE = 0.17677669529663687f;  // 1/sqrt(32)

DEV v16h cat8(v8h a, v8h b) {
  return __builtin_shufflevector(a, b, 0, 1, 2, 3, 4, 5, 6, 7,
                                       8, 9, 10, 11, 12, 13, 14, 15);
}

// reductions within 16-lane half-wave groups (xor 1,2,4,8)
DEV float rmax16(float v) {
  v = fmaxf(v, __shfl_xor(v, 1, 32));
  v = fmaxf(v, __shfl_xor(v, 2, 32));
  v = fmaxf(v, __shfl_xor(v, 4, 32));
  v = fmaxf(v, __shfl_xor(v, 8, 32));
  return v;
}
DEV float rsum16(float v) {
  v += __shfl_xor(v, 1, 32);
  v += __shfl_xor(v, 2, 32);
  v += __shfl_xor(v, 4, 32);
  v += __shfl_xor(v, 8, 32);
  return v;
}

// --- async global->LDS (per-lane 16B), ASYNCcnt-tracked --------------------
DEV void async_b128(unsigned int lds_off, const void* gptr) {
  asm volatile("global_load_async_to_lds_b128 %0, %1, off"
               :: "v"(lds_off), "v"(gptr)
               : "memory");
}
DEV void wait_async_le4() { asm volatile("s_wait_asynccnt 0x4" ::: "memory"); }
DEV void wait_async_0()   { asm volatile("s_wait_asynccnt 0x0" ::: "memory"); }
DEV void wait_ds_0()      { asm volatile("s_wait_dscnt 0x0"    ::: "memory"); }

// --- TDM: load a 32x32 f16 tile (row stride 32 elements) into LDS ----------
// D# packing per CDNA5 ISA
DEV void tdm_load_tile32x32(unsigned int lds_off, const void* gptr, int rows_left) {
  unsigned long long ga = (unsigned long long)(__SIZE_TYPE__)gptr;
  u32x4 g0;
  g0[0] = 1u;                                            // count=1, user mode
  g0[1] = lds_off;                                       // lds_addr
  g0[2] = (unsigned int)ga;                              // global_addr[31:0]
  g0[3] = (unsigned int)((ga >> 32) & 0x01FFFFFFu)       // global_addr[56:32]
          | (2u << 30);                                  // type=2 ("image")
  i32x8 g1;
  g1[0] = 0x00010000;                                    // data_size=2B
  g1[1] = (int)(32u << 16);                              // tensor_dim0=32 (lo16)
  g1[2] = (int)(((unsigned)rows_left & 0xFFFFu) << 16);  // tensor_dim1 lo16
  g1[3] = (int)((((unsigned)rows_left >> 16) & 0xFFFFu)  // tensor_dim1 hi16
                | (32u << 16));                          // tile_dim0=32
  g1[4] = 32;                                            // tile_dim1=32
  g1[5] = 32;                                            // tensor_dim0_stride=32
  g1[6] = 0;
  g1[7] = 0;
  i32x4 z4 = {};
#if __clang_major__ >= 23
  i32x8 z8 = {};
  __builtin_amdgcn_tensor_load_to_lds(g0, g1, z4, z4, z8, 0);
#else
  __builtin_amdgcn_tensor_load_to_lds(g0, g1, z4, z4, 0);
#endif
}

// ---------------------------------------------------------------------------
// 1) x = concat(flatten(x1+pos), flatten(x2+pos)) as [B,S,C] f16
// ---------------------------------------------------------------------------
__global__ void k_prep_x(const float* __restrict__ x1, const float* __restrict__ x2,
                         const float* __restrict__ pos, _Float16* __restrict__ xh) {
  int idx = blockIdx.x * blockDim.x + threadIdx.x;
  if (idx >= Bc * Sc * Cc) return;
  int ch = idx % Cc;
  int s  = (idx / Cc) % Sc;
  int b  = idx / (Cc * Sc);
  float v = (s < Np) ? x1[(b * Cc + ch) * Np + s]
                     : x2[(b * Cc + ch) * Np + (s - Np)];
  xh[idx] = (_Float16)(v + pos[ch]);
}

// ---------------------------------------------------------------------------
// weights -> f16
// ---------------------------------------------------------------------------
__global__ void k_conv_w(const float* __restrict__ wqkv, const float* __restrict__ wout,
                         _Float16* __restrict__ wqkvh, _Float16* __restrict__ wouth) {
  int idx = blockIdx.x * blockDim.x + threadIdx.x;
  if (idx < C3 * Cc) {
    wqkvh[idx] = (_Float16)wqkv[idx];
  } else if (idx < C3 * Cc + Cc * Cc) {
    int j = idx - C3 * Cc;
    wouth[j] = (_Float16)wout[j];
  }
}

// ---------------------------------------------------------------------------
// 2) QKV projection: qkv[m,n] = sum_k x[m,k]*Wqkv[n,k] + b[n]
//    One wave per 16x16 output tile; K=256 -> 8 WMMA per tile.
//    Q,K written [b,h,s,dh]; V written transposed [b,h,dh,s].
// ---------------------------------------------------------------------------
__global__ void k_qkv_gemm(const _Float16* __restrict__ xh, const _Float16* __restrict__ w,
                           const float* __restrict__ bias,
                           _Float16* __restrict__ q, _Float16* __restrict__ kt,
                           _Float16* __restrict__ vt) {
  int lane = threadIdx.x & 31;
  int wave = threadIdx.x >> 5;
  int m0 = blockIdx.x * 16;
  int n0 = (blockIdx.y * 4 + wave) * 16;
  int ln = lane & 15, hi = lane >> 4;
  int alo = hi * 8, boff = hi * 16;
  const _Float16* arow = xh + (size_t)(m0 + ln) * Cc;
  int ncol = n0 + ln;
  const _Float16* brow = w + (size_t)ncol * Cc;
  v8f acc = {};
#pragma unroll
  for (int k0 = 0; k0 < Cc; k0 += 32) {
    v16h A  = cat8(*(const v8h*)(arow + k0 + alo),
                   *(const v8h*)(arow + k0 + alo + 16));
    v16h Bm = cat8(*(const v8h*)(brow + k0 + alo),
                   *(const v8h*)(brow + k0 + alo + 16));
    acc = __builtin_amdgcn_wmma_f32_16x16x32_f16(false, A, false, Bm,
                                                 (short)0, acc, false, false);
  }
  float bv = bias[ncol];
  int t    = n0 >> 8;           // 0=Q 1=K 2=V
  int head = (n0 >> 5) & 7;
  int dh   = (n0 & 31) + ln;
  int b    = m0 / Sc;
  int sbase = m0 - b * Sc + hi * 8;
  size_t bh = (size_t)(b * NH + head);
#pragma unroll
  for (int r = 0; r < 8; ++r) {
    _Float16 vh = (_Float16)(acc[r] + bv);
    int s = sbase + r;
    if (t == 0)      q [(bh * Sc + s) * DH + dh] = vh;
    else if (t == 1) kt[(bh * Sc + s) * DH + dh] = vh;
    else             vt[(bh * DH + dh) * Sc + s] = vh;
  }
}

// ---------------------------------------------------------------------------
// 3) Flash attention: one wave per (b,head,16-query tile).
//    K chunk (32 keys x 32 dh, contiguous 2KB) staged via TDM;
//    V^T chunk (32 dh rows x 32 keys, strided) staged via async b128.
//    Double buffered; per chunk: 2 WMMA (scores) + online softmax +
//    LDS transpose of P + 2 WMMA (P x V).
// ---------------------------------------------------------------------------
__global__ void k_attn(const _Float16* __restrict__ q, const _Float16* __restrict__ kt,
                       const _Float16* __restrict__ vt, _Float16* __restrict__ ctx) {
  __shared__ __align__(128) _Float16 kbuf[2][32 * 32];
  __shared__ __align__(128) _Float16 vbuf[2][32 * 32];
  __shared__ __align__(32)  _Float16 p_lds[16 * 32];
  int lane = threadIdx.x;
  int bh = blockIdx.x / (Sc / 16);
  int qt = blockIdx.x % (Sc / 16);
  int ln = lane & 15, hi = lane >> 4;
  int alo = hi * 8, boff = hi * 16;
  const _Float16* qbase = q  + (size_t)bh * Sc * DH;
  const _Float16* kbase = kt + (size_t)bh * Sc * DH;
  const _Float16* vbase = vt + (size_t)bh * DH * Sc;

  unsigned int k_lds[2], v_lds[2];
  k_lds[0] = (unsigned int)(__SIZE_TYPE__)&kbuf[0][0];
  k_lds[1] = (unsigned int)(__SIZE_TYPE__)&kbuf[1][0];
  v_lds[0] = (unsigned int)(__SIZE_TYPE__)&vbuf[0][0];
  v_lds[1] = (unsigned int)(__SIZE_TYPE__)&vbuf[1][0];

  int vrow  = lane >> 2;        // 0..7 (row within group of 8)
  int vcol8 = (lane & 3) * 8;   // half-element offset within 32-key row

  auto issue_chunk = [&](int kc, int buf) {
    // K: one TDM op for the whole 32x32 tile (contiguous rows, stride 32)
    tdm_load_tile32x32(k_lds[buf], kbase + (size_t)kc * DH, Sc - kc);
    // V^T: 4 async b128 ops, 8 rows (dh) each, row stride Sc
#pragma unroll
    for (int j = 0; j < 4; ++j) {
      int row = j * 8 + vrow;
      unsigned int loff = v_lds[buf] + (unsigned int)(row * 32 + vcol8) * 2;
      async_b128(loff, vbase + (size_t)row * Sc + kc + vcol8);
    }
  };

  int qrow = qt * 16 + ln;
  v16h AQ = cat8(*(const v8h*)(qbase + (size_t)qrow * DH + alo),
                 *(const v8h*)(qbase + (size_t)qrow * DH + alo + 16));

  float mrow[8], lrow[8];
#pragma unroll
  for (int r = 0; r < 8; ++r) { mrow[r] = -1e30f; lrow[r] = 0.0f; }
  v8f o0 = {}, o1 = {};

  constexpr int NCH = Sc / 32;   // 49 chunks
  issue_chunk(0, 0);

  for (int i = 0; i < NCH; ++i) {
    int cur = i & 1;
    if (i + 1 < NCH) {
      wait_ds_0();                     // WAR: prior ds reads of buf done
      issue_chunk((i + 1) * 32, cur ^ 1);
      __builtin_amdgcn_s_wait_tensorcnt(1);   // chunk i's K tile landed
      wait_async_le4();                       // chunk i's V tile landed
    } else {
      __builtin_amdgcn_s_wait_tensorcnt(0);
      wait_async_0();
    }
    const _Float16* kb = &kbuf[cur][0];
    const _Float16* vb = &vbuf[cur][0];

    v16h BK0 = cat8(*(const v8h*)(kb + ln * 32 + alo),
                    *(const v8h*)(kb + ln * 32 + alo + 16));
    v16h BK1 = cat8(*(const v8h*)(kb + (16 + ln) * 32 + alo),
                    *(const v8h*)(kb + (16 + ln) * 32 + alo + 16));
    v8f z = {};
    v8f s0 = __builtin_amdgcn_wmma_f32_16x16x32_f16(false, AQ, false, BK0,
                                                    (short)0, z, false, false);
    v8f s1 = __builtin_amdgcn_wmma_f32_16x16x32_f16(false, AQ, false, BK1,
                                                    (short)0, z, false, false);
#pragma unroll
    for (int r = 0; r < 8; ++r) {
      float a  = s0[r] * ATTN_SCALE;
      float bb = s1[r] * ATTN_SCALE;
      float mx = rmax16(fmaxf(a, bb));
      float mnew = fmaxf(mrow[r], mx);
      float p0 = __expf(a - mnew);
      float p1 = __expf(bb - mnew);
      float rs = rsum16(p0 + p1);
      float corr = __expf(mrow[r] - mnew);
      lrow[r] = lrow[r] * corr + rs;
      mrow[r] = mnew;
      o0[r] *= corr;
      o1[r] *= corr;
      int prow = r + 8 * hi;                       // C-layout row of this lane
      p_lds[prow * 32 + ln]      = (_Float16)p0;   // keys kc..kc+15
      p_lds[prow * 32 + 16 + ln] = (_Float16)p1;   // keys kc+16..kc+31
    }
    __syncthreads();
    // P back in A-layout (lane = query row)
    v16h AP = cat8(*(const v8h*)(&p_lds[ln * 32 + alo]),
                   *(const v8h*)(&p_lds[ln * 32 + alo + 16]));
    // V^T rows contiguous in keys -> direct B fragments from LDS
    v16h BV0 = cat8(*(const v8h*)(vb + ln * 32 + alo),
                    *(const v8h*)(vb + ln * 32 + alo + 16));
    v16h BV1 = cat8(*(const v8h*)(vb + (16 + ln) * 32 + alo),
                    *(const v8h*)(vb + (16 + ln) * 32 + alo + 16));
    o0 = __builtin_amdgcn_wmma_f32_16x16x32_f16(false, AP, false, BV0,
                                                (short)0, o0, false, false);
    o1 = __builtin_amdgcn_wmma_f32_16x16x32_f16(false, AP, false, BV1,
                                                (short)0, o1, false, false);
    __syncthreads();
  }

  int b = bh >> 3, h = bh & 7;
#pragma unroll
  for (int r = 0; r < 8; ++r) {
    float inv = 1.0f / lrow[r];
    int s = qt * 16 + r + 8 * hi;
    size_t base = ((size_t)(b * Sc + s)) * Cc + h * DH;
    ctx[base + ln]      = (_Float16)(o0[r] * inv);
    ctx[base + 16 + ln] = (_Float16)(o1[r] * inv);
  }
}

// ---------------------------------------------------------------------------
// 4) Output projection: proj[m,n] = sum_k ctx[m,k]*Wout[n,k] + b[n]  (f32 out)
// ---------------------------------------------------------------------------
__global__ void k_out_proj(const _Float16* __restrict__ ctx, const _Float16* __restrict__ w,
                           const float* __restrict__ bias, float* __restrict__ proj) {
  int lane = threadIdx.x & 31;
  int wave = threadIdx.x >> 5;
  int m0 = blockIdx.x * 16;
  int n0 = (blockIdx.y * 4 + wave) * 16;
  int ln = lane & 15, hi = lane >> 4;
  int alo = hi * 8, boff = hi * 16;
  const _Float16* arow = ctx + (size_t)(m0 + ln) * Cc;
  int ncol = n0 + ln;
  const _Float16* brow = w + (size_t)ncol * Cc;
  v8f acc = {};
#pragma unroll
  for (int k0 = 0; k0 < Cc; k0 += 32) {
    v16h A  = cat8(*(const v8h*)(arow + k0 + alo),
                   *(const v8h*)(arow + k0 + alo + 16));
    v16h Bm = cat8(*(const v8h*)(brow + k0 + alo),
                   *(const v8h*)(brow + k0 + alo + 16));
    acc = __builtin_amdgcn_wmma_f32_16x16x32_f16(false, A, false, Bm,
                                                 (short)0, acc, false, false);
  }
  float bv = bias[ncol];
#pragma unroll
  for (int r = 0; r < 8; ++r) {
    int m = m0 + r + 8 * hi;
    proj[(size_t)m * Cc + ncol] = acc[r] + bv;
  }
}

// ---------------------------------------------------------------------------
// 5) out[b,c,pix] = proj[b, pix, c] + proj[b, N+pix, c]
// ---------------------------------------------------------------------------
__global__ void k_combine(const float* __restrict__ proj, float* __restrict__ out) {
  int idx = blockIdx.x * blockDim.x + threadIdx.x;
  if (idx >= Bc * Cc * Np) return;
  int pix = idx % Np;
  int ch  = (idx / Np) % Cc;
  int b   = idx / (Np * Cc);
  size_t base = (size_t)b * Sc * Cc;
  out[idx] = proj[base + (size_t)pix * Cc + ch] +
             proj[base + (size_t)(Np + pix) * Cc + ch];
}

// ---------------------------------------------------------------------------
extern "C" void kernel_launch(void* const* d_in, const int* in_sizes, int n_in,
                              void* d_out, int out_size, void* d_ws, size_t ws_size,
                              hipStream_t stream) {
  const float* x1   = (const float*)d_in[0];
  const float* x2   = (const float*)d_in[1];
  const float* pos  = (const float*)d_in[2];
  const float* wqkv = (const float*)d_in[3];
  const float* bqkv = (const float*)d_in[4];
  const float* wout = (const float*)d_in[5];
  const float* bout = (const float*)d_in[6];
  float* out = (float*)d_out;

  char* p = (char*)d_ws;
  auto carve = [&](size_t bytes) -> char* {
    char* r = p;
    p += (bytes + 255) & ~(size_t)255;
    return r;
  };
  _Float16* xh    = (_Float16*)carve((size_t)Bc * Sc * Cc * 2);
  _Float16* wqkvh = (_Float16*)carve((size_t)C3 * Cc * 2);
  _Float16* wouth = (_Float16*)carve((size_t)Cc * Cc * 2);
  _Float16* qws   = (_Float16*)carve((size_t)Bc * NH * Sc * DH * 2);
  _Float16* kws   = (_Float16*)carve((size_t)Bc * NH * Sc * DH * 2);
  _Float16* vws   = (_Float16*)carve((size_t)Bc * NH * Sc * DH * 2);
  _Float16* ctx   = (_Float16*)carve((size_t)Bc * Sc * Cc * 2);
  float*    proj  = (float*)   carve((size_t)Bc * Sc * Cc * 4);

  {
    int total = Bc * Sc * Cc;
    k_prep_x<<<(total + 255) / 256, 256, 0, stream>>>(x1, x2, pos, xh);
  }
  {
    int total = C3 * Cc + Cc * Cc;
    k_conv_w<<<(total + 255) / 256, 256, 0, stream>>>(wqkv, wout, wqkvh, wouth);
  }
  // M tiles = 6272/16 = 392 ; QKV N tiles = 768/16 = 48 -> 4 waves/block, y=12
  k_qkv_gemm<<<dim3((Bc * Sc) / 16, C3 / 64), 128, 0, stream>>>(
      xh, wqkvh, bqkv, qws, kws, vws);
  // (b,h) pairs * query tiles = 32 * 98 = 3136 waves
  k_attn<<<Bc * NH * (Sc / 16), 32, 0, stream>>>(qws, kws, vws, ctx);
  // out proj: N tiles = 256/16 = 16 -> y=4
  k_out_proj<<<dim3((Bc * Sc) / 16, Cc / 64), 128, 0, stream>>>(
      ctx, wouth, bout, proj);
  {
    int total = Bc * Cc * Np;
    k_combine<<<(total + 255) / 256, 256, 0, stream>>>(proj, out);
  }
}
